// MultiHeadAttentionEncoder_40492951666705
// MI455X (gfx1250) — matrix operand order repaired and checked by reading, not compiled, not run on hardware
//
#include <hip/hip_runtime.h>
#include <hip/hip_bf16.h>

#define B_  4
#define N_  2048
#define D_  512
#define H_  8
#define HD_ 64
#define NT_ (N_ / 64)

typedef __attribute__((ext_vector_type(16))) __bf16 v16bf;
typedef __attribute__((ext_vector_type(8)))  __bf16 v8bf;
typedef __attribute__((ext_vector_type(8)))  float  v8f;
typedef __attribute__((ext_vector_type(4)))  int    v4i;

#define AS1_ __attribute__((address_space(1)))
#define AS3_ __attribute__((address_space(3)))

// ---- gfx1250 async global->LDS copy (ASYNCcnt-tracked), guarded ----
#if __has_builtin(__builtin_amdgcn_global_load_async_to_lds_b128)
#define USE_ASYNC 1
__device__ __forceinline__ void cp16(void* lds, const void* g) {
    __builtin_amdgcn_global_load_async_to_lds_b128(
        (AS1_ v4i*)g, (AS3_ v4i*)lds, 0, 0);
}
#if __has_builtin(__builtin_amdgcn_s_wait_asynccnt)
#define WAIT_ASYNC() __builtin_amdgcn_s_wait_asynccnt(0)
#else
#define WAIT_ASYNC() asm volatile("s_wait_asynccnt 0" ::: "memory")
#endif
#else
#define USE_ASYNC 0
__device__ __forceinline__ void cp16(void* lds, const void* g) {
    *(float4*)lds = *(const float4*)g;     // sync fallback (reg round-trip)
}
#define WAIT_ASYNC()
#endif

// A-fragment (16x32 bf16) from a row-major [16][stride] tile.
// K = half*16 + e: two adjacent 16B chunks.
__device__ __forceinline__ v16bf load_a_frag(const __bf16* base, int stride,
                                             int m, int k0, int half) {
    const __bf16* p0 = base + (size_t)m * stride + k0 + half * 16;
    v8bf lo = *(const v8bf*)p0;
    v8bf hi = *(const v8bf*)(p0 + 8);
    return __builtin_shufflevector(lo, hi, 0,1,2,3,4,5,6,7,8,9,10,11,12,13,14,15);
}

// ---------------- pack kernels ----------------
__global__ void __launch_bounds__(256) pack_qkv_kernel(
    const float* __restrict__ q, const float* __restrict__ k, const float* __restrict__ v,
    __bf16* __restrict__ Qh, __bf16* __restrict__ Kh, __bf16* __restrict__ Vh)
{
    size_t idx = (size_t)blockIdx.x * 256 + threadIdx.x;   // over B*N*D
    int d  = (int)(idx % D_);
    size_t nd = idx / D_;
    int n  = (int)(nd % N_);
    int b  = (int)(nd / N_);
    int h  = d / HD_, hd = d % HD_;
    size_t dst = ((size_t)(h * B_ + b) * N_ + n) * HD_ + hd;
    Qh[dst] = (__bf16)(q[idx] * 0.125f);   // fold 1/sqrt(64)
    Kh[dst] = (__bf16)(k[idx]);
    Vh[dst] = (__bf16)(v[idx]);
}

__global__ void __launch_bounds__(256) pack_w_kernel(
    const float* __restrict__ W, __bf16* __restrict__ Wb)
{
    size_t idx = (size_t)blockIdx.x * 256 + threadIdx.x;   // over D*D
    Wb[idx] = (__bf16)(W[idx]);
}

// Vh [hb][N][HD] -> Vt [hb][HD][N], 64x64 LDS-tiled transpose, coalesced both ways.
__global__ void __launch_bounds__(256) transpose_v_kernel(
    const __bf16* __restrict__ Vh, __bf16* __restrict__ Vt)
{
    __shared__ __bf16 t[64][65];
    const int tid = threadIdx.x;
    const int hb  = blockIdx.y;          // H*B
    const int n0  = blockIdx.x * 64;     // N/64 tiles
    const __bf16* src = Vh + ((size_t)hb * N_ + n0) * HD_;
    #pragma unroll
    for (int i = 0; i < 2; i++) {
        int c = tid + i * 256;           // 512 chunks of 8 elems
        int n = c >> 3, j = c & 7;
        v8bf val = *(const v8bf*)(src + (size_t)n * HD_ + j * 8);
        #pragma unroll
        for (int e = 0; e < 8; e++) t[n][j * 8 + e] = val[e];
    }
    __syncthreads();
    __bf16* dst = Vt + (size_t)hb * HD_ * N_;
    #pragma unroll
    for (int i = 0; i < 2; i++) {
        int c = tid + i * 256;
        int d = c >> 3, j = c & 7;
        v8bf val;
        #pragma unroll
        for (int e = 0; e < 8; e++) val[e] = t[j * 8 + e][d];
        *(v8bf*)(dst + (size_t)d * N_ + n0 + j * 8) = val;
    }
}

// ---------------- flash attention ----------------
__global__ void __launch_bounds__(128) flash_attn_kernel(
    const __bf16* __restrict__ Qh, const __bf16* __restrict__ Kh,
    const __bf16* __restrict__ Vt, const unsigned char* __restrict__ mask,
    const float* __restrict__ bias, __bf16* __restrict__ attn_out)
{
    __shared__ __align__(16) __bf16 ldsK[2][64 * 64];      // [kn][d]
    __shared__ __align__(16) __bf16 ldsV[2][64 * 64];      // [d][kn]
    __shared__ __align__(16) float  ldsBias[2][64 * 64];   // [q_local][k_local]
    __shared__ __align__(32) __bf16 ldsP[4][16 * 64];      // per-wave P tile

    const int tid  = threadIdx.x;
    const int wave = tid >> 5;
    const int lane = tid & 31;
    const int half = lane >> 4;
    const int nn   = lane & 15;

    const int qb = blockIdx.x & (NT_ - 1);
    const int hb = blockIdx.x / NT_;
    const int h  = hb / B_;
    const int b  = hb % B_;
    const int qblk0 = qb * 64;
    const int q0 = qblk0 + wave * 16;

    const size_t headoff = (size_t)hb * N_ * HD_;
    const __bf16* qbase = Qh + headoff;
    const __bf16* kbase = Kh + headoff;
    const __bf16* vtb   = Vt + headoff;               // [HD][N]
    const float*  bblk  = bias + ((size_t)b * N_ + qblk0) * N_;

    // Q A-fragments for the two K=32 steps over HD=64
    v16bf qa0 = load_a_frag(qbase + (size_t)q0 * HD_, HD_, nn, 0,  half);
    v16bf qa1 = load_a_frag(qbase + (size_t)q0 * HD_, HD_, nn, 32, half);

    v8f o[4];
    for (int i = 0; i < 4; i++) o[i] = (v8f){0.f,0.f,0.f,0.f,0.f,0.f,0.f,0.f};
    float m_r[8], l_r[8];
    for (int r = 0; r < 8; r++) { m_r[r] = -1e30f; l_r[r] = 0.f; }

    // ---- tile issue: K (8KB) + V (8KB) + bias (16KB) via 16B async chunks ----
    auto issue_tile = [&](int kb, int buf) {
        const char* ks = (const char*)(kbase + (size_t)kb * 64 * HD_);
        #pragma unroll
        for (int i = 0; i < 4; i++) {
            int c = tid + i * 128;
            cp16((char*)ldsK[buf] + c * 16, ks + c * 16);
        }
        const __bf16* vs = vtb + kb * 64;
        #pragma unroll
        for (int i = 0; i < 4; i++) {
            int c = tid + i * 128;
            int d = c >> 3, j = c & 7;
            cp16((char*)ldsV[buf] + c * 16, (const char*)(vs + (size_t)d * N_) + j * 16);
        }
        const float* bs = bblk + kb * 64;
        #pragma unroll
        for (int i = 0; i < 8; i++) {
            int c = tid + i * 128;
            int qr = c >> 4, j = c & 15;
            cp16((char*)ldsBias[buf] + c * 16, (const char*)(bs + (size_t)qr * N_) + j * 16);
        }
    };

    issue_tile(0, 0);   // prime the pipeline

    for (int kb = 0; kb < NT_; ++kb) {
        const int cur = kb & 1;
        WAIT_ASYNC();          // this wave's async copies for buffer `cur` landed
        __syncthreads();       // all waves' copies landed; prev buffer free
        if (kb + 1 < NT_) issue_tile(kb + 1, 1 - cur);

        // S = Q*K^T : 4 column tiles of 16x16, K-reduction 64 = 2 WMMA
        v8f s[4];
        #pragma unroll
        for (int kt = 0; kt < 4; kt++) {
            v8f c = (v8f){0.f,0.f,0.f,0.f,0.f,0.f,0.f,0.f};
            v16bf b0 = *(const v16bf*)&ldsK[cur][(kt * 16 + nn) * 64 + half * 16];
            c = __builtin_amdgcn_wmma_f32_16x16x32_bf16(false, qa0, false, b0,
                                                        (short)0, c, false, false);
            v16bf b1 = *(const v16bf*)&ldsK[cur][(kt * 16 + nn) * 64 + 32 + half * 16];
            c = __builtin_amdgcn_wmma_f32_16x16x32_bf16(false, qa1, false, b1,
                                                        (short)0, c, false, false);
            s[kt] = c;
        }

        // bias (from LDS) + mask
        const int colbase = kb * 64;
        const float* bw = &ldsBias[cur][(wave * 16 + half * 8) * 64];
        #pragma unroll
        for (int kt = 0; kt < 4; kt++) {
            int col = colbase + kt * 16 + nn;
            float mk = mask[(size_t)b * N_ + col] ? -1e30f : 0.f;
            #pragma unroll
            for (int r = 0; r < 8; r++) s[kt][r] += bw[r * 64 + kt * 16 + nn] + mk;
        }

        // online softmax: row stats across the 16 lanes of each half
        float rmax[8];
        #pragma unroll
        for (int r = 0; r < 8; r++)
            rmax[r] = fmaxf(fmaxf(s[0][r], s[1][r]), fmaxf(s[2][r], s[3][r]));
        #pragma unroll
        for (int msk = 1; msk < 16; msk <<= 1)
            #pragma unroll
            for (int r = 0; r < 8; r++)
                rmax[r] = fmaxf(rmax[r], __shfl_xor(rmax[r], msk, 32));

        float corr[8], rsum[8];
        #pragma unroll
        for (int r = 0; r < 8; r++) {
            float mnew = fmaxf(m_r[r], rmax[r]);
            corr[r] = __expf(m_r[r] - mnew);
            m_r[r]  = mnew;
            rsum[r] = 0.f;
        }
        __bf16* pw = ldsP[wave];
        #pragma unroll
        for (int kt = 0; kt < 4; kt++)
            #pragma unroll
            for (int r = 0; r < 8; r++) {
                float p = __expf(s[kt][r] - m_r[r]);
                rsum[r] += p;
                pw[(half * 8 + r) * 64 + kt * 16 + nn] = (__bf16)p;
            }
        #pragma unroll
        for (int msk = 1; msk < 16; msk <<= 1)
            #pragma unroll
            for (int r = 0; r < 8; r++)
                rsum[r] += __shfl_xor(rsum[r], msk, 32);
        #pragma unroll
        for (int r = 0; r < 8; r++) l_r[r] = l_r[r] * corr[r] + rsum[r];
        #pragma unroll
        for (int nt = 0; nt < 4; nt++)
            #pragma unroll
            for (int r = 0; r < 8; r++) o[nt][r] *= corr[r];

        __syncthreads();   // publish P tile (C-layout -> A-layout round trip)

        // O += P * V
        #pragma unroll
        for (int d0 = 0; d0 < 64; d0 += 32) {
            v16bf pa = load_a_frag(pw, 64, nn, d0, half);
            #pragma unroll
            for (int nt = 0; nt < 4; nt++) {
                v16bf bv = *(const v16bf*)&ldsV[cur][(nt * 16 + nn) * 64 + d0 + half * 16];
                o[nt] = __builtin_amdgcn_wmma_f32_16x16x32_bf16(false, pa, false, bv,
                                                                (short)0, o[nt], false, false);
            }
        }
    }

    // epilogue: normalize and scatter into [B][N][D] bf16
    #pragma unroll
    for (int r = 0; r < 8; r++) {
        float inv = (l_r[r] > 0.f) ? 1.f / l_r[r] : 0.f;
        int row = q0 + half * 8 + r;
        #pragma unroll
        for (int nt = 0; nt < 4; nt++)
            attn_out[((size_t)b * N_ + row) * D_ + h * HD_ + nt * 16 + nn] =
                (__bf16)(o[nt][r] * inv);
    }
}

// ---------------- output projection: out = X @ W^T ----------------
__global__ void __launch_bounds__(128) out_proj_kernel(
    const __bf16* __restrict__ X, const __bf16* __restrict__ Wb,
    float* __restrict__ out)
{
    const int tid  = threadIdx.x;
    const int wave = tid >> 5;
    const int lane = tid & 31;
    const int half = lane >> 4;
    const int nn   = lane & 15;

    const int m0 = blockIdx.x * 64 + wave * 16;   // rows over B*N
    const int n0 = blockIdx.y * 64;               // cols over D

    v8f acc[4];
    for (int i = 0; i < 4; i++) acc[i] = (v8f){0.f,0.f,0.f,0.f,0.f,0.f,0.f,0.f};

    for (int k0 = 0; k0 < D_; k0 += 32) {
        v16bf a = load_a_frag(X + (size_t)m0 * D_, D_, nn, k0, half);
        #pragma unroll
        for (int nt = 0; nt < 4; nt++) {
            // B[K=k][N=j] = W[j][k] : contiguous in k per lane
            v16bf bfr = *(const v16bf*)&Wb[(size_t)(n0 + nt * 16 + nn) * D_ + k0 + half * 16];
            acc[nt] = __builtin_amdgcn_wmma_f32_16x16x32_bf16(false, a, false, bfr,
                                                              (short)0, acc[nt], false, false);
        }
    }
    #pragma unroll
    for (int nt = 0; nt < 4; nt++)
        #pragma unroll
        for (int r = 0; r < 8; r++)
            out[((size_t)m0 + half * 8 + r) * D_ + n0 + nt * 16 + nn] = acc[nt][r];
}

extern "C" void kernel_launch(void* const* d_in, const int* in_sizes, int n_in,
                              void* d_out, int out_size, void* d_ws, size_t ws_size,
                              hipStream_t stream) {
    const float*         q    = (const float*)d_in[0];
    const float*         k    = (const float*)d_in[1];
    const float*         v    = (const float*)d_in[2];
    const unsigned char* mask = (const unsigned char*)d_in[3];
    const float*         bias = (const float*)d_in[4];
    const float*         W    = (const float*)d_in[5];
    float* out = (float*)d_out;

    char* ws = (char*)d_ws;
    const size_t headBytes = (size_t)B_ * N_ * D_ * 2;   // 8 MiB (bf16)
    __bf16* Qh   = (__bf16*)(ws);
    __bf16* Kh   = (__bf16*)(ws + headBytes);
    __bf16* Vh   = (__bf16*)(ws + 2 * headBytes);        // row-major staging
    __bf16* Vt   = (__bf16*)(ws + 3 * headBytes);        // [hb][HD][N]
    __bf16* Attn = (__bf16*)(ws + 4 * headBytes);
    __bf16* Wb   = (__bf16*)(ws + 5 * headBytes);

    const int totalQKV = B_ * N_ * D_;                   // 4,194,304
    pack_qkv_kernel<<<totalQKV / 256, 256, 0, stream>>>(q, k, v, Qh, Kh, Vh);
    pack_w_kernel<<<(D_ * D_) / 256, 256, 0, stream>>>(W, Wb);

    dim3 tg(N_ / 64, H_ * B_);
    transpose_v_kernel<<<tg, 256, 0, stream>>>(Vh, Vt);

    flash_attn_kernel<<<H_ * B_ * NT_, 128, 0, stream>>>(Qh, Kh, Vt, mask, bias, Attn);

    dim3 pg((B_ * N_) / 64, D_ / 64);
    out_proj_kernel<<<pg, 128, 0, stream>>>(Attn, Wb, out);
}
